// ExploratoryMechanism_22204980920775
// MI455X (gfx1250) — matrix operand order repaired and checked by reading, compiled, not run on hardware
//
#include <hip/hip_runtime.h>
#include <hip/hip_bf16.h>

// ---------------------------------------------------------------------------
// Problem constants (from reference): b=4, s=2048, n=4096, d=256, TOP_N=5
// ---------------------------------------------------------------------------
#define BATCH 4
#define SEQ   2048
#define NCTX  4096
#define DIM   256
#define TOPN  5

typedef __bf16 v16bf __attribute__((ext_vector_type(16)));
typedef __bf16 v8bf  __attribute__((ext_vector_type(8)));
typedef float  v8f   __attribute__((ext_vector_type(8)));

union BF16x16 { v16bf v; v8bf h[2]; };

#define WMMA_BF16(A, B, C) \
    __builtin_amdgcn_wmma_f32_16x16x32_bf16(false, (A), false, (B), (short)0, (C), false, false)

// ---------------------------------------------------------------------------
// Kernel 1: latent projection qp = q @ Wq^T + bq  (f32 accumulate),
// emit bf16 qp, exact f32 row norms ||qp||^2, and gate = sigmoid(q.wg + bg).
// One block handles 8 query rows, 256 threads (thread = output feature e).
// ---------------------------------------------------------------------------
__global__ void __launch_bounds__(256)
em_proj_kernel(const float* __restrict__ q, const float* __restrict__ Wq,
               const float* __restrict__ bq, const float* __restrict__ wg,
               const float* __restrict__ bg,
               __bf16* __restrict__ qp_bf, float* __restrict__ q2,
               float* __restrict__ gate) {
    __shared__ float qrow[8][DIM];
    __shared__ float q2s[8];

    const int t = threadIdx.x;
    const int rowbase = blockIdx.x * 8;

    #pragma unroll
    for (int r = 0; r < 8; ++r)
        qrow[r][t] = q[(size_t)(rowbase + r) * DIM + t];
    if (t < 8) q2s[t] = 0.0f;
    __syncthreads();

    // --- projection: thread t computes output feature e = t for 8 rows ---
    float acc[8];
    #pragma unroll
    for (int r = 0; r < 8; ++r) acc[r] = 0.0f;
    const float* wrow = Wq + (size_t)t * DIM;   // Wq stored (out, in) row-major
    for (int d = 0; d < DIM; ++d) {
        const float w = wrow[d];
        #pragma unroll
        for (int r = 0; r < 8; ++r) acc[r] = fmaf(qrow[r][d], w, acc[r]);
    }
    const float bqe = bq[t];
    #pragma unroll
    for (int r = 0; r < 8; ++r) {
        const float val = acc[r] + bqe;
        qp_bf[(size_t)(rowbase + r) * DIM + t] = (__bf16)val;
        atomicAdd(&q2s[r], val * val);          // ds_add_f32
    }

    // --- gate: wave w (32 lanes) reduces dot(q_row[w], wg) ---
    const int wv = t >> 5, lane = t & 31;
    float gp = 0.0f;
    #pragma unroll
    for (int i = 0; i < 8; ++i) {
        const int d = lane * 8 + i;
        gp = fmaf(qrow[wv][d], wg[d], gp);
    }
    #pragma unroll
    for (int off = 16; off > 0; off >>= 1) gp += __shfl_down(gp, off, 32);
    if (lane == 0) {
        const float z = gp + bg[0];
        gate[rowbase + wv] = 1.0f / (1.0f + expf(-z));
    }

    __syncthreads();
    if (t < 8) q2[rowbase + t] = q2s[t];
}

// ---------------------------------------------------------------------------
// Kernel 2: bf16 conversion + exact f32 row norms for context/memory.
// One block per row of 256 elements.
// ---------------------------------------------------------------------------
__global__ void __launch_bounds__(256)
em_convert_norm_kernel(const float* __restrict__ src, __bf16* __restrict__ dst,
                       float* __restrict__ norms) {
    const int row = blockIdx.x;
    const int t   = threadIdx.x;
    const size_t off = (size_t)row * DIM + t;
    const float x = src[off];
    dst[off] = (__bf16)x;
    float p = x * x;
    #pragma unroll
    for (int o = 16; o > 0; o >>= 1) p += __shfl_down(p, o, 32);
    __shared__ float wsum[8];
    if ((t & 31) == 0) wsum[t >> 5] = p;
    __syncthreads();
    if (t == 0) {
        float s = 0.0f;
        #pragma unroll
        for (int i = 0; i < 8; ++i) s += wsum[i];
        norms[row] = s;
    }
}

// ---------------------------------------------------------------------------
// Kernel 3: fused dual-distance + gated blend + running top-5.
// Block = 128 threads (4 waves) handles 16 query rows; waves tile N by 16,
// looping n=4096 in steps of 64. K accumulation: 8 x wmma_f32_16x16x32_bf16
// per matrix (ctx, mem) -> 16 WMMAs per wave per N-step.
// ---------------------------------------------------------------------------
__global__ void __launch_bounds__(128)
em_dist_topk_kernel(const __bf16* __restrict__ qp, const __bf16* __restrict__ ctx,
                    const __bf16* __restrict__ mem, const float* __restrict__ q2,
                    const float* __restrict__ c2,  const float* __restrict__ m2,
                    const float* __restrict__ gate,
                    float* __restrict__ out_val, int* __restrict__ out_idx) {
    const int blk     = blockIdx.x;        // 0 .. BATCH*(SEQ/16)-1
    const int batch   = blk >> 7;          // / (SEQ/16)
    const int rowtile = blk & 127;
    const int t    = threadIdx.x;
    const int wave = t >> 5;
    const int lane = t & 31;
    const int lm   = lane & 15;            // lane within half-wave
    const int hi   = (lane >> 4) & 1;      // half-wave select

    const int rb = batch * SEQ + rowtile * 16;           // global row base
    const __bf16* qpb  = qp  + (size_t)rb * DIM;
    const __bf16* ctxb = ctx + (size_t)batch * NCTX * DIM;
    const __bf16* memb = mem + (size_t)batch * NCTX * DIM;
    const float*  c2b  = c2 + batch * NCTX;
    const float*  m2b  = m2 + batch * NCTX;

    // --- A fragments (16x32 bf16, contiguous K16 per half), full d=256 resident ---
    // lane (0-15|16-31) holds row M=lm; K chunks {kb..kb+7, kb+8..kb+15},
    // kb = 32f + 16*hi.
    BF16x16 afrag[8];
    {
        const __bf16* arow = qpb + (size_t)lm * DIM;
        #pragma unroll
        for (int f = 0; f < 8; ++f) {
            const int kb = f * 32 + hi * 16;
            afrag[f].h[0] = *(const v8bf*)(arow + kb);
            afrag[f].h[1] = *(const v8bf*)(arow + kb + 8);
        }
    }

    // per-lane C-tile row params: VGPR v -> row M = v + 8*hi
    float q2r[8], gr[8];
    #pragma unroll
    for (int v = 0; v < 8; ++v) {
        const int m = v + hi * 8;
        q2r[v] = q2[rb + m];
        gr[v]  = gate[rb + m];
    }

    __shared__ float dtile[16 * 65];     // padded stride 65 -> conflict-free scans
    __shared__ float mval[16][40];
    __shared__ int   midx[16][40];

    float tv[TOPN];
    int   ti[TOPN];
    #pragma unroll
    for (int p = 0; p < TOPN; ++p) { tv[p] = 3.3e38f; ti[p] = 0; }

    const int selr = t >> 3;   // selection row 0..15
    const int selj = t & 7;    // selection slot 0..7

    for (int it = 0; it < NCTX / 64; ++it) {
        const int ncol = it * 64 + wave * 16 + lm;    // this lane's B column
        const __bf16* crow = ctxb + (size_t)ncol * DIM;
        const __bf16* mrow = memb + (size_t)ncol * DIM;

        if (it + 1 < NCTX / 64) {                     // gfx1250 global_prefetch_b8
            __builtin_prefetch(crow + (size_t)64 * DIM, 0, 1);
            __builtin_prefetch(mrow + (size_t)64 * DIM, 0, 1);
        }

        v8f accC = {0.f,0.f,0.f,0.f,0.f,0.f,0.f,0.f};
        v8f accM = {0.f,0.f,0.f,0.f,0.f,0.f,0.f,0.f};
        #pragma unroll
        for (int f = 0; f < 8; ++f) {
            // B (32x16 bf16): lane holds column N=lm, contiguous K16 per half
            const int kb = f * 32 + hi * 16;
            BF16x16 bc, bm;
            bc.h[0] = *(const v8bf*)(crow + kb);
            bc.h[1] = *(const v8bf*)(crow + kb + 8);
            bm.h[0] = *(const v8bf*)(mrow + kb);
            bm.h[1] = *(const v8bf*)(mrow + kb + 8);
            accC = WMMA_BF16(afrag[f].v, bc.v, accC);
            accM = WMMA_BF16(afrag[f].v, bm.v, accM);
        }

        const float c2v = c2b[ncol];
        const float m2v = m2b[ncol];
        const int coltile = wave * 16 + lm;
        #pragma unroll
        for (int v = 0; v < 8; ++v) {
            const float dc = sqrtf(fmaxf(q2r[v] + c2v - 2.0f * accC[v], 0.0f));
            const float dm = sqrtf(fmaxf(q2r[v] + m2v - 2.0f * accM[v], 0.0f));
            const float g  = gr[v];
            dtile[(v + hi * 8) * 65 + coltile] = g * dc + (1.0f - g) * dm;
        }
        __syncthreads();

        // 8 threads/row each fold 8 candidates into a private running top-5
        #pragma unroll
        for (int k = 0; k < 8; ++k) {
            const int c = selj * 8 + k;
            const float val = dtile[selr * 65 + c];
            const int   idx = it * 64 + c;
            if (val < tv[TOPN - 1]) {
                tv[TOPN - 1] = val; ti[TOPN - 1] = idx;
                #pragma unroll
                for (int p = TOPN - 1; p > 0; --p) {
                    if (tv[p] < tv[p - 1]) {
                        const float fv = tv[p]; tv[p] = tv[p - 1]; tv[p - 1] = fv;
                        const int   iv = ti[p]; ti[p] = ti[p - 1]; ti[p - 1] = iv;
                    }
                }
            }
        }
        __syncthreads();
    }

    // --- final merge: 8x top-5 -> top-5 per row ---
    #pragma unroll
    for (int p = 0; p < TOPN; ++p) {
        mval[selr][selj * TOPN + p] = tv[p];
        midx[selr][selj * TOPN + p] = ti[p];
    }
    __syncthreads();
    if (selj == 0) {
        const int orow = (rb + selr) * TOPN;
        #pragma unroll
        for (int p = 0; p < TOPN; ++p) {
            float best = 3.4e38f; int bi = 0, bslot = 0;
            for (int u = 0; u < 8 * TOPN; ++u) {
                const float v = mval[selr][u];
                if (v < best) { best = v; bi = midx[selr][u]; bslot = u; }
            }
            mval[selr][bslot] = 3.4e38f;
            out_val[orow + p] = best;
            out_idx[orow + p] = bi;
        }
    }
}

// ---------------------------------------------------------------------------
// Launcher
// ---------------------------------------------------------------------------
extern "C" void kernel_launch(void* const* d_in, const int* in_sizes, int n_in,
                              void* d_out, int out_size, void* d_ws, size_t ws_size,
                              hipStream_t stream) {
    const float* q   = (const float*)d_in[0];   // (4, 2048, 256)
    const float* ctx = (const float*)d_in[1];   // (4, 4096, 256)
    const float* mem = (const float*)d_in[2];   // (4, 4096, 256)
    const float* Wq  = (const float*)d_in[3];   // (256, 256)
    const float* bq  = (const float*)d_in[4];   // (256,)
    const float* wg  = (const float*)d_in[5];   // (256,)
    const float* bg  = (const float*)d_in[6];   // (1,)

    // workspace layout (all 256B-aligned chunks; total ~20.2 MB)
    char* ws = (char*)d_ws;
    __bf16* qp_bf  = (__bf16*)ws;  ws += (size_t)BATCH * SEQ  * DIM * 2;   // 4 MB
    __bf16* ctx_bf = (__bf16*)ws;  ws += (size_t)BATCH * NCTX * DIM * 2;   // 8 MB
    __bf16* mem_bf = (__bf16*)ws;  ws += (size_t)BATCH * NCTX * DIM * 2;   // 8 MB
    float* q2   = (float*)ws;      ws += (size_t)BATCH * SEQ  * 4;
    float* gate = (float*)ws;      ws += (size_t)BATCH * SEQ  * 4;
    float* c2   = (float*)ws;      ws += (size_t)BATCH * NCTX * 4;
    float* m2   = (float*)ws;      ws += (size_t)BATCH * NCTX * 4;

    // 1) projection + q norms + gate
    em_proj_kernel<<<(BATCH * SEQ) / 8, 256, 0, stream>>>(
        q, Wq, bq, wg, bg, qp_bf, q2, gate);

    // 2) bf16 conversion + norms for context and memory
    em_convert_norm_kernel<<<BATCH * NCTX, 256, 0, stream>>>(ctx, ctx_bf, c2);
    em_convert_norm_kernel<<<BATCH * NCTX, 256, 0, stream>>>(mem, mem_bf, m2);

    // 3) fused WMMA distances + gated blend + top-5
    float* out_val = (float*)d_out;
    int*   out_idx = (int*)d_out + (size_t)BATCH * SEQ * TOPN;  // values then indices
    em_dist_topk_kernel<<<BATCH * (SEQ / 16), 128, 0, stream>>>(
        qp_bf, ctx_bf, mem_bf, q2, c2, m2, gate, out_val, out_idx);
}
